// MultiHeadAttention_257698037876
// MI455X (gfx1250) — matrix operand order repaired and checked by reading, not compiled, not run on hardware
//
#include <hip/hip_runtime.h>
#include <hip/hip_bf16.h>

// ---------------------------------------------------------------------------
// MHA for MI455X (gfx1250): bf16 WMMA + async global->LDS weight staging.
// ---------------------------------------------------------------------------

typedef __attribute__((ext_vector_type(16))) __bf16 v16bf;
typedef __attribute__((ext_vector_type(8)))  __bf16 v8bf;
typedef __attribute__((ext_vector_type(4)))  __bf16 v4bf;
typedef __attribute__((ext_vector_type(8)))  float  v8f;
typedef __attribute__((ext_vector_type(4)))  float  v4f;

constexpr int B_  = 2;
constexpr int S_  = 2048;
constexpr int D_  = 1024;
constexpr int H_  = 16;
constexpr int DEPTH_ = 64;

// --- operand builders ------------------------------------------------------
__device__ __forceinline__ v16bf make_a_f32(const float* __restrict__ row, int k0, int kg) {
    v4f f0 = *(const v4f*)(row + k0 + 16 * kg);
    v4f f1 = *(const v4f*)(row + k0 + 16 * kg + 4);
    v4f f2 = *(const v4f*)(row + k0 + 16 * kg + 8);
    v4f f3 = *(const v4f*)(row + k0 + 16 * kg + 12);
    v16bf r;
#pragma unroll
    for (int i = 0; i < 4; ++i) {
        r[i]      = (__bf16)f0[i];
        r[i + 4]  = (__bf16)f1[i];
        r[i + 8]  = (__bf16)f2[i];
        r[i + 12] = (__bf16)f3[i];
    }
    return r;
}
__device__ __forceinline__ v16bf make_a_bf16(const __bf16* __restrict__ row, int k0, int kg) {
    v8bf lo = *(const v8bf*)(row + k0 + 16 * kg);
    v8bf hi = *(const v8bf*)(row + k0 + 16 * kg + 8);
    return __builtin_shufflevector(lo, hi, 0, 1, 2, 3, 4, 5, 6, 7,
                                           8, 9, 10, 11, 12, 13, 14, 15);
}
__device__ __forceinline__ v16bf make_b_bf16(const __bf16* __restrict__ col, int k0, int kg) {
    return *(const v16bf*)(col + k0 + 16 * kg);
}

__device__ __forceinline__ float rmax16(float x) {
    x = fmaxf(x, __shfl_xor(x, 1, 32));
    x = fmaxf(x, __shfl_xor(x, 2, 32));
    x = fmaxf(x, __shfl_xor(x, 4, 32));
    x = fmaxf(x, __shfl_xor(x, 8, 32));
    return x;
}
__device__ __forceinline__ float rsum16(float x) {
    x += __shfl_xor(x, 1, 32);
    x += __shfl_xor(x, 2, 32);
    x += __shfl_xor(x, 4, 32);
    x += __shfl_xor(x, 8, 32);
    return x;
}

#define WMMA_BF16(a, b, c) \
    __builtin_amdgcn_wmma_f32_16x16x32_bf16(false, (a), false, (b), (short)0, (c), false, false)

// Async global->LDS 16-byte copy (CDNA5, tracked by ASYNCcnt, GV addressing).
#define ASYNC_LDS_B128(loff, gaddr) \
    asm volatile("global_load_async_to_lds_b128 %0, %1, off" :: "v"(loff), "v"(gaddr) : "memory")

// Issue one 64x32 bf16 weight tile (4 KB) into LDS buffer: 256 16B segments,
// 2 per thread across the 128-thread block (2 async ops per wave).
__device__ __forceinline__ void issue_btile(const __bf16* __restrict__ Wbf, int n0, int k0,
                                            __bf16* buf, int tid) {
#pragma unroll
    for (int j = 0; j < 2; ++j) {
        const int seg = tid * 2 + j;
        const int row = seg >> 2;
        const int q8  = (seg & 3) * 8;
        unsigned long long g = (unsigned long long)(Wbf + (size_t)(n0 + row) * D_ + k0 + q8);
        unsigned loff = (unsigned)(unsigned long long)(buf + row * 32 + q8);
        ASYNC_LDS_B128(loff, g);
    }
}

// ---------------------------------------------------------------------------
// Kernel 0: fp32 -> bf16 bulk convert (weights).
// ---------------------------------------------------------------------------
__global__ void cvt_bf16_kernel(const float* __restrict__ x, __bf16* __restrict__ y, int n) {
    int i = (blockIdx.x * blockDim.x + threadIdx.x) * 4;
    if (i < n) {
        v4f f = *(const v4f*)(x + i);
        v4bf o;
        o[0] = (__bf16)f[0]; o[1] = (__bf16)f[1];
        o[2] = (__bf16)f[2]; o[3] = (__bf16)f[3];
        *(v4bf*)(y + i) = o;
    }
}

// ---------------------------------------------------------------------------
// Kernel 1: C = X[M,K](f32) * Wbf[N,K]^T + bias, bf16 out.
// Block = 4 waves sharing one 64-wide N super-tile; B staged via async LDS
// double buffering. Wave w handles m-tile (mgroup*4 + w).
//   vmode==0: head-split store [B,H,S,DEPTH] (Q,K); vmode==1: [B,H,DEPTH,S] (V).
// ---------------------------------------------------------------------------
__global__ void proj_gemm_kernel(const float* __restrict__ X, const __bf16* __restrict__ Wbf,
                                 const float* __restrict__ bias, __bf16* __restrict__ out,
                                 int vmode) {
    __shared__ __align__(16) __bf16 bsh[2][64 * 32];
    const int tid  = threadIdx.x;
    const int lane = tid & 31;
    const int w    = tid >> 5;
    const int NT   = D_ / 64;                       // 16 N super-tiles
    const int nt   = blockIdx.x % NT;
    const int mg   = blockIdx.x / NT;               // 64 m-groups
    const int mt   = mg * 4 + w;
    const int m0   = mt * 16;
    const int n0   = nt * 64;
    const int kg   = lane >> 4;
    const int l16  = lane & 15;

    const float* arow = X + (size_t)(m0 + l16) * D_;

    issue_btile(Wbf, n0, 0, bsh[0], tid);
    int p = 0;
    v8f acc[4] = {};
    for (int k0 = 0; k0 < D_; k0 += 32) {
        if (k0 + 32 < D_) {
            issue_btile(Wbf, n0, k0 + 32, bsh[p ^ 1], tid);
            asm volatile("s_wait_asynccnt 0x2" ::: "memory");   // my 2 oldest landed
        } else {
            asm volatile("s_wait_asynccnt 0x0" ::: "memory");
        }
        __syncthreads();                                        // whole tile landed
        __builtin_prefetch(arow + k0 + 64, 0, 1);
        v16bf a = make_a_f32(arow, k0, kg);
#pragma unroll
        for (int t = 0; t < 4; ++t) {
            v16bf b = *(const v16bf*)(&bsh[p][(t * 16 + l16) * 32 + 16 * kg]);
            acc[t] = WMMA_BF16(a, b, acc[t]);
        }
        __syncthreads();                                        // safe to overwrite buf p
        p ^= 1;
    }

    const int bb = m0 / S_;
#pragma unroll
    for (int t = 0; t < 4; ++t) {
        const int n  = n0 + t * 16 + l16;
        const int h  = n >> 6;
        const int d  = n & 63;
        const float bv = bias[n];
#pragma unroll
        for (int r = 0; r < 8; ++r) {
            const int m = m0 + 8 * kg + r;
            const int s = m - bb * S_;
            const float v = acc[t][r] + bv;
            size_t idx;
            if (vmode) idx = ((size_t)((bb * H_ + h) * DEPTH_ + d)) * S_ + s;
            else       idx = ((size_t)((bb * H_ + h) * S_ + s)) * DEPTH_ + d;
            out[idx] = (__bf16)v;
        }
    }
}

// ---------------------------------------------------------------------------
// Kernel 2: causal scores + softmax (per-wave 16-row q stripe; online m/l,
// recompute pass writes normalized probs; zero-fill masked region).
// ---------------------------------------------------------------------------
__global__ void attn_softmax_kernel(const __bf16* __restrict__ Q, const __bf16* __restrict__ K,
                                    float* __restrict__ attn) {
    const int lane = threadIdx.x & 31;
    const int wid  = blockIdx.x * (blockDim.x >> 5) + (threadIdx.x >> 5);
    const int QT   = S_ / 16;
    const int bh   = wid / QT;
    const int qt   = wid % QT;
    if (bh >= B_ * H_) return;
    const int kg  = lane >> 4;
    const int l16 = lane & 15;

    const __bf16* qrow = Q + ((size_t)bh * S_ + qt * 16 + l16) * DEPTH_;
    const v16bf a0 = make_a_bf16(qrow, 0, kg);
    const v16bf a1 = make_a_bf16(qrow, 32, kg);
    const float scale = 0.125f;

    float m[8], l[8];
#pragma unroll
    for (int r = 0; r < 8; ++r) { m[r] = -3.0e38f; l[r] = 0.0f; }

    for (int kt = 0; kt <= qt; ++kt) {
        const __bf16* kcol = K + ((size_t)bh * S_ + kt * 16 + l16) * DEPTH_;
        v16bf b0 = make_b_bf16(kcol, 0, kg);
        v16bf b1 = make_b_bf16(kcol, 32, kg);
        v8f c = {};
        c = WMMA_BF16(a0, b0, c);
        c = WMMA_BF16(a1, b1, c);
        const int col = kt * 16 + l16;
#pragma unroll
        for (int r = 0; r < 8; ++r) {
            const int row = qt * 16 + 8 * kg + r;
            float s = c[r] * scale;
            if (col > row) s = -1.0e9f;
            const float tmax = rmax16(s);
            const float mn   = fmaxf(m[r], tmax);
            const float e    = __expf(s - mn);
            const float es   = rsum16(e);
            l[r] = l[r] * __expf(m[r] - mn) + es;
            m[r] = mn;
        }
    }
    float linv[8];
#pragma unroll
    for (int r = 0; r < 8; ++r) linv[r] = __frcp_rn(l[r]);

    for (int kt = 0; kt <= qt; ++kt) {
        const __bf16* kcol = K + ((size_t)bh * S_ + kt * 16 + l16) * DEPTH_;
        v16bf b0 = make_b_bf16(kcol, 0, kg);
        v16bf b1 = make_b_bf16(kcol, 32, kg);
        v8f c = {};
        c = WMMA_BF16(a0, b0, c);
        c = WMMA_BF16(a1, b1, c);
        const int col = kt * 16 + l16;
#pragma unroll
        for (int r = 0; r < 8; ++r) {
            const int row = qt * 16 + 8 * kg + r;
            float s = c[r] * scale;
            if (col > row) s = -1.0e9f;
            const float pr = __expf(s - m[r]) * linv[r];
            attn[((size_t)bh * S_ + row) * S_ + col] = pr;
        }
    }
    const int start = (qt + 1) * 16;
    if (start < S_) {
        const int n4 = (S_ - start) >> 2;
        const float4 z = make_float4(0.f, 0.f, 0.f, 0.f);
        for (int i = lane; i < 16 * n4; i += 32) {
            const int r  = i / n4;
            const int c4 = i - r * n4;
            float* p = attn + ((size_t)bh * S_ + qt * 16 + r) * S_ + start + c4 * 4;
            *reinterpret_cast<float4*>(p) = z;
        }
    }
}

// ---------------------------------------------------------------------------
// Kernel 3: head_out = P @ V with V pre-transposed ([B,H,DEPTH,S] bf16).
// ---------------------------------------------------------------------------
__global__ void attn_v_kernel(const float* __restrict__ attn, const __bf16* __restrict__ Vt,
                              __bf16* __restrict__ hout) {
    const int lane = threadIdx.x & 31;
    const int wid  = blockIdx.x * (blockDim.x >> 5) + (threadIdx.x >> 5);
    const int QT   = S_ / 16;
    const int bh   = wid / QT;
    const int qt   = wid % QT;
    if (bh >= B_ * H_) return;
    const int kg  = lane >> 4;
    const int l16 = lane & 15;

    const float* arow = attn + ((size_t)bh * S_ + qt * 16 + l16) * S_;
    const __bf16* vcol[4];
#pragma unroll
    for (int t = 0; t < 4; ++t) vcol[t] = Vt + ((size_t)bh * DEPTH_ + t * 16 + l16) * S_;

    v8f acc[4] = {};
    int kend = ((qt + 1) * 16 + 31) & ~31;
    if (kend > S_) kend = S_;
    for (int k0 = 0; k0 < kend; k0 += 32) {
        __builtin_prefetch(arow + k0 + 64, 0, 1);
        v16bf a = make_a_f32(arow, k0, kg);
#pragma unroll
        for (int t = 0; t < 4; ++t) {
            v16bf b = make_b_bf16(vcol[t], k0, kg);
            acc[t] = WMMA_BF16(a, b, acc[t]);
        }
    }
    const int bb = bh / H_;
    const int h  = bh - bb * H_;
#pragma unroll
    for (int t = 0; t < 4; ++t) {
        const int d = t * 16 + l16;
#pragma unroll
        for (int r = 0; r < 8; ++r) {
            const int s = qt * 16 + 8 * kg + r;
            hout[((size_t)(bb * S_ + s)) * D_ + h * 64 + d] = (__bf16)acc[t][r];
        }
    }
}

// ---------------------------------------------------------------------------
// Kernel 4: out = Hout[M,K](bf16) * Wbf[N,K]^T + dense_b, fp32 out [M,D].
// Same async-LDS-staged structure as the projection GEMM.
// ---------------------------------------------------------------------------
__global__ void dense_kernel(const __bf16* __restrict__ Xbf, const __bf16* __restrict__ Wbf,
                             const float* __restrict__ bias, float* __restrict__ out) {
    __shared__ __align__(16) __bf16 bsh[2][64 * 32];
    const int tid  = threadIdx.x;
    const int lane = tid & 31;
    const int w    = tid >> 5;
    const int NT   = D_ / 64;
    const int nt   = blockIdx.x % NT;
    const int mg   = blockIdx.x / NT;
    const int mt   = mg * 4 + w;
    const int m0   = mt * 16;
    const int n0   = nt * 64;
    const int kg   = lane >> 4;
    const int l16  = lane & 15;

    const __bf16* arow = Xbf + (size_t)(m0 + l16) * D_;

    issue_btile(Wbf, n0, 0, bsh[0], tid);
    int p = 0;
    v8f acc[4] = {};
    for (int k0 = 0; k0 < D_; k0 += 32) {
        if (k0 + 32 < D_) {
            issue_btile(Wbf, n0, k0 + 32, bsh[p ^ 1], tid);
            asm volatile("s_wait_asynccnt 0x2" ::: "memory");
        } else {
            asm volatile("s_wait_asynccnt 0x0" ::: "memory");
        }
        __syncthreads();
        __builtin_prefetch(arow + k0 + 64, 0, 1);
        v16bf a = make_a_bf16(arow, k0, kg);
#pragma unroll
        for (int t = 0; t < 4; ++t) {
            v16bf b = *(const v16bf*)(&bsh[p][(t * 16 + l16) * 32 + 16 * kg]);
            acc[t] = WMMA_BF16(a, b, acc[t]);
        }
        __syncthreads();
        p ^= 1;
    }
#pragma unroll
    for (int t = 0; t < 4; ++t) {
        const int n  = n0 + t * 16 + l16;
        const float bv = bias[n];
#pragma unroll
        for (int r = 0; r < 8; ++r) {
            const int m = m0 + 8 * kg + r;
            out[(size_t)m * D_ + n] = acc[t][r] + bv;
        }
    }
}

// ---------------------------------------------------------------------------
extern "C" void kernel_launch(void* const* d_in, const int* in_sizes, int n_in,
                              void* d_out, int out_size, void* d_ws, size_t ws_size,
                              hipStream_t stream) {
    const float* v    = (const float*)d_in[0];
    const float* k    = (const float*)d_in[1];
    const float* q    = (const float*)d_in[2];
    // d_in[3] = causal mask (known analytically; unused)
    const float* wq_w = (const float*)d_in[4];
    const float* wq_b = (const float*)d_in[5];
    const float* wk_w = (const float*)d_in[6];
    const float* wk_b = (const float*)d_in[7];
    const float* wv_w = (const float*)d_in[8];
    const float* wv_b = (const float*)d_in[9];
    const float* dw   = (const float*)d_in[10];
    const float* db   = (const float*)d_in[11];

    float* out  = (float*)d_out;                       // [B,S,D]
    float* attn = out + (size_t)B_ * S_ * D_;          // [B,H,S,S]

    const size_t NBH = (size_t)B_ * H_ * S_ * DEPTH_;  // 4,194,304 elems
    const size_t NW  = (size_t)D_ * D_;                // 1,048,576 elems
    __bf16* qbf  = (__bf16*)d_ws;
    __bf16* kbf  = qbf  + NBH;
    __bf16* vtbf = kbf  + NBH;
    __bf16* hbf  = vtbf + NBH;
    __bf16* wqbf = hbf  + NBH;
    __bf16* wkbf = wqbf + NW;
    __bf16* wvbf = wkbf + NW;
    __bf16* dwbf = wvbf + NW;

    const dim3 blk(128);                                   // 4 waves / block
    const int gemm_blocks = ((B_ * S_) / 16 / 4) * (D_ / 64);  // 64 * 16 = 1024
    const int attn_waves  = B_ * H_ * (S_ / 16);               // 4096
    const int cvt_blocks  = (int)(NW / 4 / 256);               // 1024

    cvt_bf16_kernel<<<cvt_blocks, 256, 0, stream>>>(wq_w, wqbf, (int)NW);
    cvt_bf16_kernel<<<cvt_blocks, 256, 0, stream>>>(wk_w, wkbf, (int)NW);
    cvt_bf16_kernel<<<cvt_blocks, 256, 0, stream>>>(wv_w, wvbf, (int)NW);
    cvt_bf16_kernel<<<cvt_blocks, 256, 0, stream>>>(dw,   dwbf, (int)NW);

    proj_gemm_kernel<<<gemm_blocks, blk, 0, stream>>>(q, wqbf, wq_b, qbf, 0);
    proj_gemm_kernel<<<gemm_blocks, blk, 0, stream>>>(k, wkbf, wk_b, kbf, 0);
    proj_gemm_kernel<<<gemm_blocks, blk, 0, stream>>>(v, wvbf, wv_b, vtbf, 1);
    attn_softmax_kernel<<<attn_waves / 4, blk, 0, stream>>>(qbf, kbf, attn);
    attn_v_kernel<<<attn_waves / 4, blk, 0, stream>>>(attn, vtbf, hbf);
    dense_kernel<<<gemm_blocks, blk, 0, stream>>>(hbf, dwbf, db, out);
}
